// MoESlotDecoder_28458453303802
// MI455X (gfx1250) — matrix operand order repaired and checked by reading, not compiled, not run on hardware
//
#include <hip/hip_runtime.h>
#include <hip/hip_bf16.h>

// ---------------------------------------------------------------------------
// MoESlotDecoder fused forward for MI455X (gfx1250, wave32, WMMA + TDM).
//
// Shapes: B=4096, S=64, D=512, C=4, K(top)=16, T=0.01
// One workgroup (256 thr = 8 wave32) per batch b:
//   slots[b] (64x512 f32, 128KB) -> LDS via TENSOR_LOAD_TO_LDS (TDM),
//                                   waited with s_wait_tensorcnt 0.
//   GEMM1: H = relu(X @ W1^T + b1)   v_wmma_f32_16x16x32_bf16 (A cvt f32->bf16
//                                    at fragment build; B streamed from L2)
//   GEMM2: S = H @ W2^T + b2         (S aliases dead X staging in LDS)
//   GEMM3: keep = relu(S @ Wk1^T + bk1) . Wk2 fused via ds_add_f32
//   softmax(keep/T), top-16 hard mask, gate-normalize, 64x4 decode, outputs.
// Weights converted once to bf16 into d_ws (L2-resident, 1.5MB).
// Roofline: 512MB slots read @ 23.3TB/s ~ 22us dominates; 412 GFLOP of bf16
// WMMA and ~6GB of L2 weight re-reads fit well underneath.
// ---------------------------------------------------------------------------

typedef __attribute__((ext_vector_type(16))) __bf16 v16bf;
typedef __attribute__((ext_vector_type(8)))  __bf16 v8bf;
typedef __attribute__((ext_vector_type(4)))  __bf16 v4bf;
typedef __attribute__((ext_vector_type(8)))  float  v8f;
typedef __attribute__((ext_vector_type(4)))  unsigned int v4u;
typedef __attribute__((ext_vector_type(8)))  int    v8i;
typedef __attribute__((ext_vector_type(4)))  int    v4i;

#define DIMD   512
#define NSLOT  64
#define NCLS   4
#define NBATCH 4096
#define TOPK   16

#if defined(__AMDGCN__) && __has_builtin(__builtin_amdgcn_tensor_load_to_lds)
#define MOE_HAVE_TDM 1
#else
#define MOE_HAVE_TDM 0
#endif

union ABFrag { v16bf v; v8bf h[2]; };

// ---------------------------------------------------------------------------
// Weight fp32 -> bf16 conversion (once per launch; tiny vs main kernel).
// Layout in ws: W1bf [0,256K elts), W2bf [256K,512K), Wk1bf [512K,768K).
// ---------------------------------------------------------------------------
__global__ __launch_bounds__(256) void moe_prep_weights(
    const float* __restrict__ W1, const float* __restrict__ W2,
    const float* __restrict__ Wk1, __bf16* __restrict__ ws)
{
    const int i = blockIdx.x * 256 + threadIdx.x;      // 0 .. 786431
    if (i < 262144)       ws[i] = (__bf16)W1[i];
    else if (i < 524288)  ws[i] = (__bf16)W2[i - 262144];
    else                  ws[i] = (__bf16)Wk1[i - 524288];
}

// 8-element fragment chunk loaders (LDS): bf16 direct, f32 with convert.
__device__ __forceinline__ v8bf load8(const __bf16* p) {
    return *(const v8bf*)p;
}
__device__ __forceinline__ v8bf load8(const float* p) {
    v8bf r;
#pragma unroll
    for (int i = 0; i < 8; ++i) r[i] = (__bf16)p[i];
    return r;
}

// ---------------------------------------------------------------------------
// One 64x512 x 512x512 GEMM slice per wave:
//   Out[m, n] = act( sum_k A[m,k] * Bw[n,k] + bias[n] )
// A in LDS (bf16 or f32, row stride 512), Bw global bf16 [out,in] row-major.
// A fragments for all 16 K-steps cached in VGPRs across the 16 N-tiles.
// ---------------------------------------------------------------------------
template <bool RELU, typename AT>
__device__ __forceinline__ void gemm_tile(
    const AT* __restrict__ A, const __bf16* __restrict__ Bw,
    const float* __restrict__ bias, __bf16* __restrict__ Out,
    int mtile, int nbase, int lane)
{
    // ISA 16-bit A layout: lane L holds row m=L%16; halves 0-7 = K at
    // koff (0 for lanes<16, 8 for lanes>=16), halves 8-15 = K at koff+16.
    v16bf a[16];
    const int arow = mtile * 16 + (lane & 15);
    const int koff = (lane >> 4) << 3;               // 0 or 8
    const AT* abase = A + (size_t)arow * DIMD + koff;
#pragma unroll
    for (int k = 0; k < 16; ++k) {
        ABFrag t;
        t.h[0] = load8(abase + k * 32);
        t.h[1] = load8(abase + k * 32 + 16);
        a[k] = t.v;
    }

    const int rbase = mtile * 16 + ((lane >> 4) << 3);   // C rows: v + 0/8
    for (int nt = 0; nt < 16; ++nt) {
        const int ncol  = (nbase + nt) * 16 + (lane & 15);
        // B layout: lane = column n; halves 0-7 = K at koff, halves 8-15 =
        // K at koff+16 (koff 0 or 8 by lane group), matching A.
        const __bf16* bbase = Bw + (size_t)ncol * DIMD + ((lane >> 4) << 3);
        if (nt + 1 < 16)   // warm near caches for next weight panel
            __builtin_prefetch(Bw + (size_t)((nbase + nt + 1) * 16 + (lane & 15)) * DIMD, 0, 3);
        v8f c = {};
#pragma unroll
        for (int k = 0; k < 16; ++k) {
            ABFrag t;
            t.h[0] = *(const v8bf*)(bbase + k * 32);
            t.h[1] = *(const v8bf*)(bbase + k * 32 + 16);
            c = __builtin_amdgcn_wmma_f32_16x16x32_bf16(
                    false, a[k], false, t.v, (short)0, c, false, false);
        }
        const float bs = bias[ncol];
#pragma unroll
        for (int v = 0; v < 8; ++v) {
            float val = c[v] + bs;
            if (RELU) val = fmaxf(val, 0.0f);
            Out[(size_t)(rbase + v) * DIMD + ncol] = (__bf16)val;
        }
    }
}

// ---------------------------------------------------------------------------
// Main fused kernel: one block per batch element.
// Dynamic LDS: Xf32 128KB (TDM target; aliased by Sbf) | Hbf 64KB | tail
// ---------------------------------------------------------------------------
__global__ __launch_bounds__(256) void moe_fused_forward(
    const float* __restrict__ slots,
    const float* __restrict__ b1,  const float* __restrict__ b2,
    const float* __restrict__ Wd,  const float* __restrict__ bd,
    const float* __restrict__ bk1, const float* __restrict__ Wk2,
    const float* __restrict__ bk2,
    const __bf16* __restrict__ wbf,
    float* __restrict__ out)
{
    extern __shared__ char smem[];
    float*  Xf   = (float*)smem;                  // 128 KB staging (LDS off 0)
    __bf16* Hbf  = (__bf16*)(smem + 131072);      // 64 KB
    __bf16* Sbf  = (__bf16*)smem;                 // aliases dead Xf after GEMM1
    float*  ksm  = (float*)(smem + 196608);       // 64 keep scores
    float*  gate = ksm + NSLOT;                   // 64 unnormalized gates
    float*  hrd  = gate + NSLOT;                  // 64 hard mask
    float*  xacc = hrd + NSLOT;                   // 4 class accumulators

    const int b    = blockIdx.x;
    const int tid  = threadIdx.x;
    const int lane = tid & 31;
    const int wave = tid >> 5;

    // ---- Phase 0: slots[b] (128KB f32) HBM -> LDS ----
#if MOE_HAVE_TDM
    if (tid == 0) {
        // Tensor DMA descriptor: 2-D tensor, 4-byte elems, 512x64 tile,
        // row stride 512, destination LDS offset 0.
        const unsigned long long ga =
            (unsigned long long)(size_t)(slots + (size_t)b * NSLOT * DIMD);
        v4u g0;
        g0.x = 1u;                                   // count=1, user desc
        g0.y = 0u;                                   // lds_addr = 0
        g0.z = (unsigned int)ga;                     // global_addr[31:0]
        g0.w = (unsigned int)((ga >> 32) & 0x1FFFFFFull) | (2u << 30); // type=2
        v8i g1;
        g1[0] = (int)(2u << 16);                     // data_size = 4B
        g1[1] = (int)(512u << 16);                   // tensor_dim0 = 512 (lo16)
        g1[2] = (int)(64u << 16);                    // tensor_dim1 = 64  (lo16)
        g1[3] = (int)(512u << 16);                   // tile_dim0 = 512
        g1[4] = 64;                                  // tile_dim1 = 64
        g1[5] = 512;                                 // tensor_dim0_stride
        g1[6] = 0;
        g1[7] = 0;
        v4i gz = {0, 0, 0, 0};
#if defined(__clang_major__) && (__clang_major__ >= 23)
        v8i gz8 = {0, 0, 0, 0, 0, 0, 0, 0};
        __builtin_amdgcn_tensor_load_to_lds(g0, g1, gz, gz, gz8, 0);
#else
        __builtin_amdgcn_tensor_load_to_lds(g0, g1, gz, gz, 0);
#endif
        __builtin_amdgcn_s_wait_tensorcnt(0);
    }
#else
    {
        const float4* src = (const float4*)(slots + (size_t)b * NSLOT * DIMD);
        float4* dst = (float4*)Xf;
        for (int i = tid; i < (NSLOT * DIMD) / 4; i += 256) dst[i] = src[i];
    }
#endif
    if (tid < NSLOT) ksm[tid] = bk2[0];
    if (tid < NCLS)  xacc[tid] = 0.0f;
    __syncthreads();

    const int mtile = wave & 3;          // 4 M-tiles of 16 rows
    const int nbase = (wave >> 2) * 16;  // waves 0-3: n-tiles 0-15; 4-7: 16-31

    // ---- GEMM1: H = relu(X @ W1^T + b1)  (A = f32 LDS, cvt in frag build) ----
    gemm_tile<true >(Xf,  wbf,          b1, Hbf, mtile, nbase, lane);
    __syncthreads();
    // ---- GEMM2: S = H @ W2^T + b2  (S overwrites dead X staging) ----
    gemm_tile<false>(Hbf, wbf + 262144, b2, Sbf, mtile, nbase, lane);
    __syncthreads();

    // ---- GEMM3 fused: keep[r] += relu(S@Wk1^T + bk1)[r,n] * Wk2[n] ----
    {
        const __bf16* Bw = wbf + 524288;        // Wk1 bf16
        v16bf a[16];
        const int arow = mtile * 16 + (lane & 15);
        const int koff = (lane >> 4) << 3;
        const __bf16* abase = Sbf + (size_t)arow * DIMD + koff;
#pragma unroll
        for (int k = 0; k < 16; ++k) {
            ABFrag t;
            t.h[0] = load8(abase + k * 32);
            t.h[1] = load8(abase + k * 32 + 16);
            a[k] = t.v;
        }
        const int rbase = mtile * 16 + ((lane >> 4) << 3);
        for (int nt = 0; nt < 16; ++nt) {
            const int ncol = (nbase + nt) * 16 + (lane & 15);
            const __bf16* bbase = Bw + (size_t)ncol * DIMD + ((lane >> 4) << 3);
            v8f c = {};
#pragma unroll
            for (int k = 0; k < 16; ++k) {
                ABFrag t;
                t.h[0] = *(const v8bf*)(bbase + k * 32);
                t.h[1] = *(const v8bf*)(bbase + k * 32 + 16);
                c = __builtin_amdgcn_wmma_f32_16x16x32_bf16(
                        false, a[k], false, t.v, (short)0, c, false, false);
            }
            const float bs = bk1[ncol];
            const float ws = Wk2[ncol];
#pragma unroll
            for (int v = 0; v < 8; ++v) {
                float hk = fmaxf(c[v] + bs, 0.0f);
                atomicAdd(&ksm[rbase + v], hk * ws);   // ds_add_f32
            }
        }
    }
    __syncthreads();

    // ---- softmax(keep/T), top-16 hard mask (rank with index tie-break) ----
    if (tid < NSLOT) {
        const float my = ksm[tid];
        float mx = -3.0e38f;
        for (int j = 0; j < NSLOT; ++j) mx = fmaxf(mx, ksm[j]);
        float sum = 0.0f;
        for (int j = 0; j < NSLOT; ++j) sum += __expf((ksm[j] - mx) * 100.0f);
        const float soft = __expf((my - mx) * 100.0f) / sum;
        int rank = 0;
        for (int j = 0; j < NSLOT; ++j) {
            const float o = ksm[j];
            rank += (o > my) || (o == my && j < tid);
        }
        const float hard = (rank < TOPK) ? 1.0f : 0.0f;
        gate[tid] = soft * hard;
        hrd[tid]  = hard;
    }
    __syncthreads();

    // ---- gate normalization + write slot_gate / hard_keep ----
    if (tid < NSLOT) {
        float gs = 1e-8f;
        for (int j = 0; j < NSLOT; ++j) gs += gate[j];
        out[(size_t)NBATCH * NCLS + (size_t)b * NSLOT + tid] = gate[tid] / gs;
        out[(size_t)NBATCH * NCLS + (size_t)NBATCH * NSLOT
            + (size_t)b * NSLOT + tid] = hrd[tid];
    }

    // ---- x[b,c] = sum_r gate_n[r] * (S[r,:].Wd[c,:] + bd[c]) ----
    {
        const int r = tid >> 2, c = tid & 3;
        float gs = 1e-8f;
        for (int j = 0; j < NSLOT; ++j) gs += gate[j];
        const __bf16* srow = Sbf + (size_t)r * DIMD;
        const float*  wrow = Wd + (size_t)c * DIMD;
        float dot = 0.0f;
#pragma unroll 8
        for (int d = 0; d < DIMD; ++d) dot += (float)srow[d] * wrow[d];
        atomicAdd(&xacc[c], (gate[r] / gs) * (dot + bd[c]));
    }
    __syncthreads();
    if (tid < NCLS) out[(size_t)b * NCLS + tid] = xacc[tid];
}

// ---------------------------------------------------------------------------
extern "C" void kernel_launch(void* const* d_in, const int* in_sizes, int n_in,
                              void* d_out, int out_size, void* d_ws, size_t ws_size,
                              hipStream_t stream) {
    const float* slots = (const float*)d_in[0];
    const float* W1    = (const float*)d_in[1];
    const float* b1    = (const float*)d_in[2];
    const float* W2    = (const float*)d_in[3];
    const float* b2    = (const float*)d_in[4];
    const float* Wd    = (const float*)d_in[5];
    const float* bd    = (const float*)d_in[6];
    const float* Wk1   = (const float*)d_in[7];
    const float* bk1   = (const float*)d_in[8];
    const float* Wk2   = (const float*)d_in[9];
    const float* bk2   = (const float*)d_in[10];
    float* out  = (float*)d_out;
    __bf16* wbf = (__bf16*)d_ws;     // 3 * 512*512 bf16 = 1.5 MB

    moe_prep_weights<<<3072, 256, 0, stream>>>(W1, W2, Wk1, wbf);

    const size_t lds_bytes = 196608 + 1024;   // X staging + H tile + tail
    moe_fused_forward<<<NBATCH, 256, lds_bytes, stream>>>(
        slots, b1, b2, Wd, bd, bk1, Wk2, bk2, wbf, out);
}
